// SetAbstraction_5145370821372
// MI455X (gfx1250) — compile-verified, matrix operand order checked by reading, not run on hardware
//
#include <hip/hip_runtime.h>
#include <float.h>

// ---------------------------------------------------------------------------
// Problem constants (from reference)
// ---------------------------------------------------------------------------
constexpr int B = 16;
constexpr int N = 4096;
constexpr int M = 1024;
constexpr int C = 64;
constexpr int KNN = 32;                    // k of ball query
constexpr long R_ROWS = (long)B * M * KNN; // 524288 samples for BN stats
constexpr float RADIUS2 = 0.15f * 0.15f;
constexpr float BN_EPS = 1e-5f;
constexpr int OUTC = 128;
constexpr long QOFF = (long)B * M * 3;     // q occupies front of d_out

// wave32 / WGP friendly
constexpr int TPB = 256;                   // 8 waves
constexpr int TILE_ROWS = 128;             // 8 waves x 16 rows

typedef __bf16 bf16_t;
typedef __bf16 v16bf __attribute__((ext_vector_type(16)));
typedef float  v8f   __attribute__((ext_vector_type(8)));

// ---------------------------------------------------------------------------
// Workspace layout (bytes). Activations kept in bf16: exactly the WMMA operand
// precision, halves HBM traffic vs fp32 (the pipeline is bandwidth bound:
// ~17 GFLOP vs ~200MB of activation traffic at 23.3 TB/s -> ~10us floor).
// ---------------------------------------------------------------------------
constexpr size_t OFF_IDX   = 0;                                   // int32 [B*M*K]        2 MB
constexpr size_t OFF_Y0    = OFF_IDX + (size_t)R_ROWS * 4;        // bf16 [R*64]         64 MB
constexpr size_t OFF_Y1    = OFF_Y0  + (size_t)R_ROWS * 64 * 2;   // bf16 [R*64]         64 MB
constexpr size_t OFF_XT    = OFF_Y1  + (size_t)R_ROWS * 64 * 2;   // f32  [B*N*64]       16 MB
constexpr size_t OFF_PMAX  = OFF_XT  + (size_t)B * N * C * 4;     // f32  [B*M*128]       8 MB
constexpr size_t OFF_PMIN  = OFF_PMAX + (size_t)B * M * OUTC * 4; // f32  [B*M*128]       8 MB
constexpr size_t OFF_STATS = OFF_PMIN + (size_t)B * M * OUTC * 4; // f32  [512]
// stats layout: sum0[64] sq0[64] sum1[64] sq1[64] sum2[128] sq2[128]

// ---------------------------------------------------------------------------
// Float atomic max/min via integer ordering trick (lowers to
// global_atomic_max_i32 / global_atomic_min_u32 — no return needed).
// Valid for buffers initialized to -inf / +inf.
// ---------------------------------------------------------------------------
__device__ __forceinline__ void atomicMaxF(float* a, float v) {
    if (v >= 0.0f) atomicMax((int*)a, __float_as_int(v));
    else           atomicMin((unsigned int*)a, __float_as_uint(v));
}
__device__ __forceinline__ void atomicMinF(float* a, float v) {
    if (v >= 0.0f) atomicMin((int*)a, __float_as_int(v));
    else           atomicMax((unsigned int*)a, __float_as_uint(v));
}

// ---------------------------------------------------------------------------
// K0: re-initialize mutable ws state every call (stats=0, pooled = -+inf).
// ---------------------------------------------------------------------------
__global__ void init_kernel(float* stats, float* pmax, float* pmin) {
    long t = (long)blockIdx.x * TPB + threadIdx.x;
    if (t < 512) stats[t] = 0.0f;
    if (t < (long)B * M * OUTC) {
        pmax[t] = -FLT_MAX;
        pmin[t] =  FLT_MAX;
    }
}

// ---------------------------------------------------------------------------
// K1: transpose x [B,C,N] -> xT [B,N,C] so point-gathers are 64 contiguous
// floats (x is L2-resident at 16.8MB/192MB L2, so the scatter side is cheap).
// ---------------------------------------------------------------------------
__global__ void xpose_kernel(const float* __restrict__ x, float* __restrict__ xT) {
    long t = (long)blockIdx.x * TPB + threadIdx.x;   // coalesced read of x
    int n = (int)(t & (N - 1));
    int c = (int)((t >> 12) & (C - 1));
    int b = (int)(t >> 18);
    xT[(((long)b << 12) + n) * C + c] = x[t];
}

// ---------------------------------------------------------------------------
// K2: ball query. One wave (32 lanes) per query; p[b] staged in 48KB LDS.
// Ordered first-K selection via __ballot + prefix popcount (lane order ==
// point order, so selection matches the argsort-stable reference).
// ---------------------------------------------------------------------------
__global__ void ballquery_kernel(const float* __restrict__ p,
                                 const float* __restrict__ q,
                                 int* __restrict__ idx) {
    __shared__ float sp[N * 3];
    const int b    = blockIdx.x / (M / 8);
    const int mblk = blockIdx.x % (M / 8);
    const float* pb = p + (size_t)b * N * 3;
    for (int t = threadIdx.x; t < N * 3; t += TPB) sp[t] = pb[t];
    __syncthreads();

    const int wave = threadIdx.x >> 5;
    const int lane = threadIdx.x & 31;
    const int m = mblk * 8 + wave;
    const float* qp = q + ((size_t)b * M + m) * 3;
    const float qx = qp[0], qy = qp[1], qz = qp[2];
    int* out = idx + ((size_t)b * M + m) * KNN;

    int cnt = 0;
    int firstIdx = 0;
    bool haveFirst = false;
    for (int base = 0; base < N && cnt < KNN; base += 32) {
        const int j = base + lane;
        const float dx = sp[j * 3 + 0] - qx;
        const float dy = sp[j * 3 + 1] - qy;
        const float dz = sp[j * 3 + 2] - qz;
        const bool pred = (dx * dx + dy * dy + dz * dz) < RADIUS2;
        const unsigned mask = (unsigned)__ballot(pred);
        if (!haveFirst && mask) { firstIdx = base + __ffs(mask) - 1; haveFirst = true; }
        if (pred) {
            const int slot = cnt + __popc(mask & ((1u << lane) - 1u));
            if (slot < KNN) out[slot] = j;
        }
        cnt += __popc(mask);
    }
    if (cnt < KNN)
        for (int s = cnt + lane; s < KNN; s += 32) out[s] = firstIdx; // pad (0 if none)
}

// ---------------------------------------------------------------------------
// GEMM + BN-stat kernel, templated per layer.
//   MODE 0: A = gathered [p_hat | x] features (CIN=67, zero-padded to 96)
//   MODE 1: A = BN(prev stats)+ReLU of yin (CIN=64)
//   MODE 2: like MODE 1, COUT=128, no y store; pooled max/min over k instead.
// One WG = 8 waves; each wave owns 16 rows x COUT via v_wmma_f32_16x16x32_bf16.
// A and B are packed in LDS in exact WMMA bf16 fragment order so each lane's
// fragment is one contiguous 32B ds_load_b128 pair.
// Fragment mapping (16-bit A 16x32, ISA 7.12.2): lane = half*16 + m,
//   element e<8  -> K = e + 8*half ;  e>=8 -> K = (e-8) + 16 + 8*half.
// Epilogue bounces D through (dead) A-tile LDS so global y stores are
// coalesced b128 instead of per-lane scattered b16.
// ---------------------------------------------------------------------------
template <int CIN, int KS, int COUT, int MODE>
__global__ void gemm_bn_kernel(const float* __restrict__ p,
                               const float* __restrict__ q,
                               const float* __restrict__ xT,
                               const int*   __restrict__ idx,
                               const bf16_t* __restrict__ yin,
                               const float* __restrict__ inSum,
                               const float* __restrict__ inSq,
                               const float* __restrict__ inG,
                               const float* __restrict__ inB,
                               const float* __restrict__ W,
                               bf16_t* __restrict__ yout,
                               float* __restrict__ outSum,
                               float* __restrict__ outSq,
                               float* __restrict__ pMax,
                               float* __restrict__ pMin) {
    constexpr int NT = COUT / 16;
    constexpr int KPAD = KS * 32;

    __shared__ __align__(32) bf16_t s_a[8 * KS * 32 * 16]; // also reused as D bounce
    __shared__ __align__(32) bf16_t s_b[NT * KS * 32 * 16];
    __shared__ float s_scale[64], s_shift[64];
    __shared__ float s_sum[COUT], s_sq[COUT];

    const int tid  = threadIdx.x;
    const int lane = tid & 31;
    const int wave = tid >> 5;
    const long tileBase = (long)blockIdx.x * TILE_ROWS;

    if (MODE >= 1) {
        // warm the incoming activation tile (128 rows x 64 bf16 = 16KB) while
        // we stage weights: global_prefetch_b8, one line per thread.
        __builtin_prefetch((const void*)(yin + (tileBase << 6) + tid * 32), 0, 3);
    }

    if (tid < COUT) { s_sum[tid] = 0.0f; s_sq[tid] = 0.0f; }
    if (MODE >= 1 && tid < CIN) {
        // finalize producer BN stats (redundant per WG — 64 rsqrts, free)
        const float inv = 1.0f / (float)R_ROWS;
        const float mean = inSum[tid] * inv;
        const float var  = inSq[tid] * inv - mean * mean;
        const float sc   = inG[tid] * rsqrtf(var + BN_EPS);
        s_scale[tid] = sc;
        s_shift[tid] = inB[tid] - mean * sc;
    }
    __syncthreads();

    // ---- stage B (weights, zero-padded) into WMMA fragment layout ----
    for (int t = tid; t < COUT * KPAD; t += TPB) {
        const int o = t / KPAD;
        const int c = t % KPAD;
        const float wv = (c < CIN) ? W[(long)o * CIN + c] : 0.0f;
        const int ks = c >> 5, kk = c & 31;
        const int half = (kk >> 3) & 1;
        const int e = (kk & 7) | ((kk >> 4) << 3);
        const int nt = o >> 4, n = o & 15;
        s_b[(((nt * KS + ks) * 32) + half * 16 + n) * 16 + e] = (bf16_t)wv;
    }

    // ---- stage A (128 rows) into WMMA fragment layout ----
    for (int t = tid; t < TILE_ROWS * KPAD; t += TPB) {
        const int row = t / KPAD;
        const int c   = t % KPAD;
        const long rg = tileBase + row;
        float v = 0.0f;
        if (MODE == 0) {
            const int bb = (int)(rg >> 15);            // / (M*KNN)
            const int mm = (int)((rg >> 5) & (M - 1));
            const int j  = idx[rg];
            if (c < 3) {
                v = p[((size_t)bb * N + j) * 3 + c] - q[((size_t)bb * M + mm) * 3 + c];
            } else if (c < 3 + C) {
                v = xT[(((size_t)bb * N + j) << 6) + (c - 3)];
            }
        } else {
            if (c < CIN) {
                const float yv = (float)yin[(rg << 6) + c];
                v = fmaxf(0.0f, s_scale[c] * yv + s_shift[c]); // BN + ReLU fused
            }
        }
        const int ks = c >> 5, kk = c & 31;
        const int half = (kk >> 3) & 1;
        const int e = (kk & 7) | ((kk >> 4) << 3);
        const int mrow = row & 15, wv2 = row >> 4;
        s_a[(((wv2 * KS + ks) * 32) + half * 16 + mrow) * 16 + e] = (bf16_t)v;
    }
    __syncthreads();

    // ---- WMMA main loop: acc[nt] += A(16x32) * B(32x16) ----
    const v16bf* av = (const v16bf*)s_a;
    const v16bf* bv = (const v16bf*)s_b;
    v8f acc[NT] = {};
#pragma unroll
    for (int ks = 0; ks < KS; ++ks) {
        const v16bf a = av[(wave * KS + ks) * 32 + lane];
#pragma unroll
        for (int nt = 0; nt < NT; ++nt) {
            acc[nt] = __builtin_amdgcn_wmma_f32_16x16x32_bf16(
                false, a, false, bv[(nt * KS + ks) * 32 + lane],
                (short)0, acc[nt], false, false);
        }
    }
    __syncthreads(); // all fragment reads done: A region may be reused for D

    // ---- epilogue: stats, pooled max/min, D -> LDS in [row][col] order ----
    bf16_t* s_out = s_a; // 8*KS*512 >= 128*64 elements for KS>=2
    const int ncol = lane & 15;
    const int halfsel = lane >> 4; // C/D layout: lane<16 rows m..m+7, else +8
#pragma unroll
    for (int nt = 0; nt < NT; ++nt) {
        const int col = nt * 16 + ncol;
        float sum = 0.0f, sq = 0.0f, mx = -FLT_MAX, mn = FLT_MAX;
#pragma unroll
        for (int i = 0; i < 8; ++i) {
            const float v = acc[nt][i];
            sum += v; sq += v * v;
            mx = fmaxf(mx, v); mn = fminf(mn, v);
            if (MODE < 2) {
                const int row = wave * 16 + halfsel * 8 + i;
                s_out[row * 64 + col] = (bf16_t)v;
            }
        }
        atomicAdd(&s_sum[col], sum);   // ds_add_f32
        atomicAdd(&s_sq[col], sq);
        if (MODE == 2) {
            // all 16 rows of this wave's tile share one (b,m) group (K=32)
            const float mx2 = fmaxf(mx, __shfl_xor(mx, 16));
            const float mn2 = fminf(mn, __shfl_xor(mn, 16));
            if (halfsel == 0) {
                const long grp = (tileBase + wave * 16) >> 5; // b*M + m
                atomicMaxF(&pMax[grp * OUTC + col], mx2);
                atomicMinF(&pMin[grp * OUTC + col], mn2);
            }
        }
    }
    __syncthreads();

    if (MODE < 2) {
        // coalesced tile store: 128x64 bf16 = 16KB as b128 per thread (x4)
        const uint4* src = (const uint4*)s_out;
        uint4* dst = (uint4*)(yout + (tileBase << 6));
        for (int t = tid; t < TILE_ROWS * 64 / 8; t += TPB) dst[t] = src[t];
    }
    if (tid < COUT) {
        atomicAdd(&outSum[tid], s_sum[tid]); // one global atomic per ch per WG
        atomicAdd(&outSq[tid], s_sq[tid]);
    }
}

// ---------------------------------------------------------------------------
// K6: finalize — BN2 + ReLU on pooled pre-activation extrema. relu(a*y+c) is
// monotone in y: pick pooled max when scale>=0, pooled min when scale<0, so
// pooling pre-BN is exact. Writes out [B,128,M].
// ---------------------------------------------------------------------------
__global__ void finalize_kernel(const float* __restrict__ stats,
                                const float* __restrict__ g2,
                                const float* __restrict__ b2,
                                const float* __restrict__ pMax,
                                const float* __restrict__ pMin,
                                float* __restrict__ out) {
    const long t = (long)blockIdx.x * TPB + threadIdx.x;
    const int m  = (int)(t & (M - 1));
    const int ch = (int)((t >> 10) & (OUTC - 1));
    const int b  = (int)(t >> 17);
    const float* sum2 = stats + 256;
    const float* sq2  = stats + 384;
    const float inv = 1.0f / (float)R_ROWS;
    const float mean = sum2[ch] * inv;
    const float var  = sq2[ch] * inv - mean * mean;
    const float sc   = g2[ch] * rsqrtf(var + BN_EPS);
    const float sh   = b2[ch] - mean * sc;
    const size_t pi = ((size_t)b * M + m) * OUTC + ch;
    const float v = (sc >= 0.0f) ? pMax[pi] : pMin[pi];
    out[QOFF + t] = fmaxf(0.0f, sc * v + sh);
}

__global__ void copyq_kernel(const float* __restrict__ q, float* __restrict__ out) {
    const long t = (long)blockIdx.x * TPB + threadIdx.x;
    if (t < QOFF) out[t] = q[t];
}

// ---------------------------------------------------------------------------
// Launch pipeline (everything on `stream`; no allocations, graph-capture safe)
// ---------------------------------------------------------------------------
extern "C" void kernel_launch(void* const* d_in, const int* in_sizes, int n_in,
                              void* d_out, int out_size, void* d_ws, size_t ws_size,
                              hipStream_t stream) {
    const float* p  = (const float*)d_in[0];
    const float* q  = (const float*)d_in[1];
    const float* x  = (const float*)d_in[2];
    const float* W0 = (const float*)d_in[3];
    const float* g0 = (const float*)d_in[4];
    const float* b0 = (const float*)d_in[5];
    const float* W1 = (const float*)d_in[6];
    const float* g1 = (const float*)d_in[7];
    const float* b1 = (const float*)d_in[8];
    const float* W2 = (const float*)d_in[9];
    const float* g2 = (const float*)d_in[10];
    const float* b2 = (const float*)d_in[11];
    float* out = (float*)d_out;

    char* ws = (char*)d_ws;
    int*    idx   = (int*)(ws + OFF_IDX);
    bf16_t* y0    = (bf16_t*)(ws + OFF_Y0);
    bf16_t* y1    = (bf16_t*)(ws + OFF_Y1);
    float*  xT    = (float*)(ws + OFF_XT);
    float*  pMax  = (float*)(ws + OFF_PMAX);
    float*  pMin  = (float*)(ws + OFF_PMIN);
    float*  stats = (float*)(ws + OFF_STATS);
    float* sum0 = stats, *sq0 = stats + 64;
    float* sum1 = stats + 128, *sq1 = stats + 192;
    float* sum2 = stats + 256, *sq2 = stats + 384;

    // 0) reset per-call state (stats, pooled extrema)
    init_kernel<<<(B * M * OUTC) / TPB, TPB, 0, stream>>>(stats, pMax, pMin);
    // 1) x [B,C,N] -> xT [B,N,C]
    xpose_kernel<<<(B * C * N) / TPB, TPB, 0, stream>>>(x, xT);
    // 2) ball query
    ballquery_kernel<<<B * (M / 8), TPB, 0, stream>>>(p, q, idx);

    const int gemmBlocks = (int)(R_ROWS / TILE_ROWS); // 4096
    // 3) layer0: gather [p_hat|x] (67 -> pad 96) x W0 -> y0[.,64] + stats0
    gemm_bn_kernel<67, 3, 64, 0><<<gemmBlocks, TPB, 0, stream>>>(
        p, q, xT, idx, nullptr, nullptr, nullptr, nullptr, nullptr,
        W0, y0, sum0, sq0, nullptr, nullptr);
    // 4) layer1: BN0+ReLU(y0) x W1 -> y1[.,64] + stats1
    gemm_bn_kernel<64, 2, 64, 1><<<gemmBlocks, TPB, 0, stream>>>(
        nullptr, nullptr, nullptr, nullptr, y0, sum0, sq0, g0, b0,
        W1, y1, sum1, sq1, nullptr, nullptr);
    // 5) layer2: BN1+ReLU(y1) x W2 -> pooled max/min over k + stats2
    gemm_bn_kernel<64, 2, 128, 2><<<gemmBlocks, TPB, 0, stream>>>(
        nullptr, nullptr, nullptr, nullptr, y1, sum1, sq1, g1, b1,
        W2, nullptr, sum2, sq2, pMax, pMin);

    // 6) BN2 + ReLU on pooled extrema -> out[B,128,M]; copy q
    finalize_kernel<<<(B * OUTC * M) / TPB, TPB, 0, stream>>>(
        stats, g2, b2, pMax, pMin, out);
    copyq_kernel<<<(int)((QOFF + TPB - 1) / TPB), TPB, 0, stream>>>(q, out);
}
